// RNN_31267361915158
// MI455X (gfx1250) — matrix operand order repaired and checked by reading, not compiled, not run on hardware
//
#include <hip/hip_runtime.h>
#include <hip/hip_bf16.h>

// Problem constants (match reference)
#define S_LEN 512
#define BATCH 128
#define IN_D  256
#define HID   512
#define OUT_D 128

typedef __attribute__((ext_vector_type(16))) __bf16 v16bf;
typedef __attribute__((ext_vector_type(8)))  float  v8f;

// ---------------------------------------------------------------------------
// WMMA helpers (CDNA5 gfx1250, wave32)
// ---------------------------------------------------------------------------
__device__ __forceinline__ v8f wmma_bf16(v16bf a, v16bf b, v8f c) {
    // D = A(16x32 bf16) * B(32x16 bf16) + C(16x16 f32)
    return __builtin_amdgcn_wmma_f32_16x16x32_bf16(
        /*neg_a=*/false, a, /*neg_b=*/false, b,
        /*c_mod=*/(short)0, c, /*reuse_a=*/false, /*reuse_b=*/false);
}

// A-fragment loader from row-major bf16 [M x K] tile (ld in elements).
// Per ISA: lane L holds row r=L&15; lanes 0-15 carry K = 0..7,16..23;
// lanes 16-31 carry K = 8..15,24..31.  Two contiguous 16B chunks per lane.
__device__ __forceinline__ v16bf load_A_bf16(const __bf16* base, int ld, int lane) {
    int r   = lane & 15;
    int hiK = (lane >> 4) << 3;                  // 0 or 8
    const __bf16* p = base + (size_t)r * ld + hiK;
    union { uint4 q[2]; v16bf v; } u;
    u.q[0] = *(const uint4*)(p);                 // K = hiK .. hiK+7
    u.q[1] = *(const uint4*)(p + 16);            // K = 16+hiK .. 16+hiK+7
    return u.v;
}

// A-fragment loader from row-major f32 tile with on-the-fly bf16 convert.
__device__ __forceinline__ v16bf load_A_f32(const float* base, int ld, int lane) {
    int r   = lane & 15;
    int hiK = (lane >> 4) << 3;
    const float* p = base + (size_t)r * ld + hiK;
    union { v16bf v; __bf16 e[16]; } u;
#pragma unroll
    for (int i = 0; i < 8; ++i) {
        u.e[i]     = (__bf16)p[i];
        u.e[8 + i] = (__bf16)p[16 + i];
    }
    return u.v;
}

// Read a prebuilt B fragment (32 lanes x 16 bf16) from LDS.
__device__ __forceinline__ v16bf lds_frag(const __bf16* p) {
    union { uint4 q[2]; v16bf v; } u;
    u.q[0] = *(const uint4*)(p);
    u.q[1] = *(const uint4*)(p + 8);
    return u.v;
}

// Build one B fragment (k0..k0+31 rows, n0..n0+15 cols of row-major f32 [K x N])
// into LDS, converting f32 -> bf16.  Per ISA B layout: lane = col (&15),
// lanes 0-15 hold K 0..7,16..23; lanes 16-31 hold K 8..15,24..31; element i = K (kb + i + (i & 8)).
__device__ __forceinline__ void build_B_frag(__bf16* dst /*[32][16]*/,
                                             const float* W, int ldw,
                                             int k0, int n0, int lane) {
    int c  = lane & 15;
    int kb = (lane >> 4) << 3;                   // 0 or 8
    const float* p = W + (size_t)(k0 + kb) * ldw + n0 + c;
    __bf16* d = dst + lane * 16;
#pragma unroll
    for (int i = 0; i < 16; ++i) d[i] = (__bf16)p[(size_t)(i + (i & 8)) * ldw];
}

// ---------------------------------------------------------------------------
// Init: zero h_0 slot and global barrier counter
// ---------------------------------------------------------------------------
__global__ void rnn_init_kernel(__bf16* h0, unsigned* bar, int n) {
    int i = blockIdx.x * blockDim.x + threadIdx.x;
    if (i == 0) *bar = 0u;
    if (i < n) h0[i] = (__bf16)0.0f;
}

// ---------------------------------------------------------------------------
// Phase 1: xh[s,b,:] = x[s,b,:] @ W_xh + b_h        (65536 x 512, K=256)
// Block = 256 threads (8 waves); WG computes a 128-row x 16-col strip.
// W_xh tile (256x16) staged in LDS as 8 B-fragments shared by all waves.
// ---------------------------------------------------------------------------
__global__ void rnn_xh_gemm(const float* __restrict__ x,
                            const float* __restrict__ Wxh,
                            const float* __restrict__ bh,
                            float* __restrict__ xh) {
    __shared__ __align__(16) __bf16 Bf[8][32][16];     // 8 KB
    const int wave = threadIdx.x >> 5, lane = threadIdx.x & 31;
    const int n0 = blockIdx.y * 16;

    // wave w builds k-fragment w (K chunk w*32..w*32+31)
    build_B_frag(&Bf[wave][0][0], Wxh, HID, wave * 32, n0, lane);
    __syncthreads();

    const size_t m0 = (size_t)blockIdx.x * 128 + (size_t)wave * 16;
    v8f acc = {};
#pragma unroll
    for (int kt = 0; kt < IN_D / 32; ++kt) {
        v16bf a = load_A_f32(x + m0 * IN_D + kt * 32, IN_D, lane);
        v16bf b = lds_frag(&Bf[kt][lane][0]);
        acc = wmma_bf16(a, b, acc);
    }
    const int c = lane & 15, rb = (lane >> 4) << 3;
    const float bias = bh[n0 + c];
#pragma unroll
    for (int i = 0; i < 8; ++i)
        xh[(m0 + rb + i) * HID + n0 + c] = acc[i] + bias;
}

// ---------------------------------------------------------------------------
// Phase 2: persistent recurrence.  Grid = 8 WGs, each owns a 64-wide slice of
// the hidden dimension.  W_hh's slice lives in LDS as 64 prebuilt bf16 B-frags
// (64 KB), reused for all 512 steps.  h history kept in bf16 (hbuf slot t),
// step t computes slot t+1.  Global atomic barrier between steps.
// Epilogue: stage f32 accumulators to LDS (68-float pitch, conflict-free),
// re-read row-contiguously, add xh (coalesced b128), tanh, pack bf16, and
// store h_next with coalesced global_store_b128.
// ---------------------------------------------------------------------------
#define CPITCH 68   // 64 cols + 4 pad floats -> 2-way-max LDS bank conflicts

__global__ void rnn_recurrence(const float* __restrict__ Whh,
                               const float* __restrict__ xh,
                               __bf16* __restrict__ hbuf,
                               unsigned* __restrict__ bar) {
    __shared__ __align__(16) __bf16 Bf[16][4][32][16];     // 64 KB B-fragments
    __shared__ __align__(16) float  Cstage[8][16][CPITCH]; // ~34 KB epilogue staging
    const int wave = threadIdx.x >> 5, lane = threadIdx.x & 31;
    const int n_base = blockIdx.x * 64;

    // Build W_hh B-fragments once: 64 frags / 8 waves = 8 each.
    for (int f = wave; f < 64; f += 8) {
        int kt = f >> 2, nt = f & 3;
        build_B_frag(&Bf[kt][nt][0][0], Whh, HID, kt * 32, n_base + nt * 16, lane);
    }
    __syncthreads();

    const int m_base = wave * 16;                 // wave owns batch rows m_base..+15
    const int c = lane & 15, rb = (lane >> 4) << 3;
    const int r2 = lane >> 1;                     // epilogue: row per lane-pair
    const int ch = (lane & 1) * 32;               // epilogue: 32-col half
    const unsigned nwg = gridDim.x;

    for (int t = 0; t < S_LEN; ++t) {
        const __bf16* hprev = hbuf + (size_t)t * (BATCH * HID);
        v8f acc[4] = {};
#pragma unroll
        for (int kt = 0; kt < HID / 32; ++kt) {
            v16bf a = load_A_bf16(hprev + (size_t)m_base * HID + kt * 32, HID, lane);
#pragma unroll
            for (int nt = 0; nt < 4; ++nt)
                acc[nt] = wmma_bf16(a, lds_frag(&Bf[kt][nt][lane][0]), acc[nt]);
        }

        if (t + 1 < S_LEN)   // prefetch next step's xh strip (global_prefetch_b8)
            __builtin_prefetch(xh + ((size_t)(t + 1) * BATCH + m_base) * HID + n_base, 0, 1);

        // ---- epilogue: stage raw accumulators to LDS (C layout: lane = col) ----
        float* sc = &Cstage[wave][0][0];
#pragma unroll
        for (int nt = 0; nt < 4; ++nt)
#pragma unroll
            for (int i = 0; i < 8; ++i)
                sc[(rb + i) * CPITCH + nt * 16 + c] = acc[nt][i];
        __syncthreads();   // cross-lane LDS visibility before re-read

        // re-read one contiguous 32-float row-half per lane; fuse +xh, tanh, cvt
        const float* srow = sc + r2 * CPITCH + ch;
        const float* xr   = xh + ((size_t)t * BATCH + m_base + r2) * HID + n_base + ch;
        __bf16* hp = hbuf + (size_t)(t + 1) * (BATCH * HID)
                   + (size_t)(m_base + r2) * HID + n_base + ch;
        union { uint4 q[4]; __bf16 e[32]; } o;
#pragma unroll
        for (int j = 0; j < 32; ++j)
            o.e[j] = (__bf16)tanhf(srow[j] + xr[j]);
#pragma unroll
        for (int j = 0; j < 4; ++j)
            *(uint4*)(hp + j * 8) = o.q[j];      // 4x global_store_b128, coalesced

        // device-wide barrier: all 8 WGs finish step t before step t+1 reads h
        __threadfence();
        __syncthreads();
        if (threadIdx.x == 0) {
            __hip_atomic_fetch_add(bar, 1u, __ATOMIC_ACQ_REL, __HIP_MEMORY_SCOPE_AGENT);
            const unsigned target = (unsigned)(t + 1) * nwg;
            while (__hip_atomic_load(bar, __ATOMIC_ACQUIRE, __HIP_MEMORY_SCOPE_AGENT) < target)
                __builtin_amdgcn_s_sleep(1);
        }
        __syncthreads();
    }
}

// ---------------------------------------------------------------------------
// Phase 3: out = hs @ W_hy + b_y      (65536 x 128, K=512)
// hs = bf16 history slots 1..S (contiguous).  W_hy tile staged as 16 B-frags.
// ---------------------------------------------------------------------------
__global__ void rnn_out_gemm(const __bf16* __restrict__ hs,
                             const float* __restrict__ Why,
                             const float* __restrict__ by,
                             float* __restrict__ out) {
    __shared__ __align__(16) __bf16 Bf[16][32][16];    // 16 KB
    const int wave = threadIdx.x >> 5, lane = threadIdx.x & 31;
    const int n0 = blockIdx.y * 16;

    for (int kt = wave; kt < 16; kt += 8)
        build_B_frag(&Bf[kt][0][0], Why, OUT_D, kt * 32, n0, lane);
    __syncthreads();

    const size_t m0 = (size_t)blockIdx.x * 128 + (size_t)wave * 16;
    v8f acc = {};
#pragma unroll
    for (int kt = 0; kt < HID / 32; ++kt) {
        v16bf a = load_A_bf16(hs + m0 * HID + kt * 32, HID, lane);
        acc = wmma_bf16(a, lds_frag(&Bf[kt][lane][0]), acc);
    }
    const int c = lane & 15, rb = (lane >> 4) << 3;
    const float bias = by[n0 + c];
#pragma unroll
    for (int i = 0; i < 8; ++i)
        out[(m0 + rb + i) * OUT_D + n0 + c] = acc[i] + bias;
}

// ---------------------------------------------------------------------------
// Host-side launcher
// ---------------------------------------------------------------------------
extern "C" void kernel_launch(void* const* d_in, const int* in_sizes, int n_in,
                              void* d_out, int out_size, void* d_ws, size_t ws_size,
                              hipStream_t stream) {
    const float* x   = (const float*)d_in[0];  // [S,B,I]
    const float* Wxh = (const float*)d_in[1];  // [I,H]
    const float* Whh = (const float*)d_in[2];  // [H,H]
    const float* Why = (const float*)d_in[3];  // [H,O]
    const float* bh  = (const float*)d_in[4];  // [H]
    const float* by  = (const float*)d_in[5];  // [O]
    float* out = (float*)d_out;                // [S,B,O]

    // Workspace layout
    const size_t XH_BYTES   = (size_t)S_LEN * BATCH * HID * sizeof(float);        // 128 MB
    const size_t HBUF_BYTES = (size_t)(S_LEN + 1) * BATCH * HID * sizeof(__bf16); // ~64 MB
    char* ws = (char*)d_ws;
    float*    xh   = (float*)ws;
    __bf16*   hbuf = (__bf16*)(ws + XH_BYTES);              // slot 0 = h_{-1} = 0
    unsigned* bar  = (unsigned*)(ws + XH_BYTES + HBUF_BYTES);

    // 0) zero h0 slot + barrier counter (must run every launch: graph-replay safe)
    rnn_init_kernel<<<dim3((BATCH * HID + 255) / 256), dim3(256), 0, stream>>>(
        hbuf, bar, BATCH * HID);

    // 1) xh = x @ W_xh + b_h
    rnn_xh_gemm<<<dim3((S_LEN * BATCH) / 128, HID / 16), dim3(256), 0, stream>>>(
        x, Wxh, bh, xh);

    // 2) sequential recurrence, persistent grid of 8 WGs (one 64-col slice each)
    rnn_recurrence<<<dim3(HID / 64), dim3(256), 0, stream>>>(Whh, xh, hbuf, bar);

    // 3) out = hs @ W_hy + b_y   (hs = hbuf slots 1..S)
    rnn_out_gemm<<<dim3((S_LEN * BATCH) / 128, OUT_D / 16), dim3(256), 0, stream>>>(
        hbuf + (size_t)BATCH * HID, Why, by, out);
}
